// DuelingDQN_90477781058029
// MI455X (gfx1250) — matrix operand order repaired and checked by reading, not compiled, not run on hardware
//
#include <hip/hip_runtime.h>

// ---------------------------------------------------------------------------
// DuelingDQN LSTM stack on gfx1250.
//  - bf16 WMMA (v_wmma_f32_16x16x32_bf16), f32 accumulation
//  - 128 blocks x 256 threads (8 waves); each block owns 32 batch rows for
//    all 100 timesteps; wave w owns hidden cols [32w,32w+32) of every layer
//  - h0/h1 ping-pong in LDS (bf16), c-state entirely in VGPRs
//  - weights pre-packed (prologue kernel) into per-lane WMMA B-fragment order
//  - next-timestep x tile prefetched via Tensor Data Mover (TENSORcnt) when
//    the builtin is available; plain loads otherwise
// ---------------------------------------------------------------------------

typedef unsigned short u16;
typedef __attribute__((ext_vector_type(16))) __bf16        v16bf;
typedef __attribute__((ext_vector_type(8)))  float         v8f;
typedef __attribute__((ext_vector_type(4)))  unsigned int  uint32x4;
typedef __attribute__((ext_vector_type(4)))  int           int32x4;
typedef __attribute__((ext_vector_type(8)))  int           int32x8;

#define B_TOT   4096
#define T_LEN   100
#define IN_DIM  140
#define IN_PAD  160
#define HID     256
#define TILE_ELEMS 512   // packed 16x32 bf16 weight tile = 1KB

#if defined(__has_builtin)
#  if __has_builtin(__builtin_amdgcn_tensor_load_to_lds) && \
      __has_builtin(__builtin_amdgcn_s_wait_tensorcnt)
#    define USE_TDM 1
#  endif
#endif
#ifndef USE_TDM
#  define USE_TDM 0
#endif

__device__ __forceinline__ u16 f2bf(float f) {
  unsigned u = __float_as_uint(f);
  unsigned r = (u + 0x7FFFu + ((u >> 16) & 1u)) >> 16;
  return (u16)r;
}
__device__ __forceinline__ float bf2f(u16 h) {
  return __uint_as_float(((unsigned)h) << 16);
}
__device__ __forceinline__ float sigm(float x) { return 1.f / (1.f + __expf(-x)); }
__device__ __forceinline__ float tanh_(float x) {
  float e = __expf(2.f * x);
  return (e - 1.f) / (e + 1.f);
}

__device__ __forceinline__ v8f wmma_bf16(v16bf a, v16bf b, v8f c) {
  return __builtin_amdgcn_wmma_f32_16x16x32_bf16(
      false, a, false, b, (short)0, c, false, false);
}

// A fragment (16x32 bf16, MxK) from LDS row-major (element stride).
__device__ __forceinline__ v16bf load_a(const u16* base, int stride,
                                        int rowBase, int k0, int m, int kh) {
  const u16* p = base + (rowBase + m) * stride + k0 + kh * 8;
  union { uint4 u[2]; v16bf v; } f;
  f.u[0] = *(const uint4*)p;
  f.u[1] = *(const uint4*)(p + 16);
  return f.v;
}

// B fragment (32x16 bf16, KxN) from pre-packed global weights: 32 contiguous
// bytes per lane (two clause'd global_load_b128, L2-resident).
__device__ __forceinline__ v16bf load_b(const u16* wp, int tile, int lane) {
  const uint4* p = (const uint4*)(wp + ((size_t)tile << 9) + lane * 16);
  union { uint4 u[2]; v16bf v; } f;
  f.u[0] = p[0];
  f.u[1] = p[1];
  return f.v;
}

#if USE_TDM
// TDM descriptor for one 32-row x 140-col f32 tile of x (row stride T*IN).
// D# group0: count=1 | lds_addr | global_addr | type=2.  group1: data_size=4B,
// tensor_dim0=140, tensor_dim1=32, tile_dim0=140, tile_dim1=32,
// tensor_dim0_stride=14000 elements.
__device__ __forceinline__ void tdm_load_x(const float* gsrc, unsigned ldsByteAddr) {
  unsigned long long ga = (unsigned long long)(uintptr_t)gsrc;
  uint32x4 g0 = { 1u, ldsByteAddr, (unsigned)ga,
                  (unsigned)((ga >> 32) & 0x01FFFFFFull) | (2u << 30) };
  int32x8 g1 = { (int)(2u << 16),    // data_size code 2 (4 bytes)
                 (int)(140u << 16),  // tensor_dim0 low16 @ [31:16]
                 (int)(32u << 16),   // tensor_dim1 low16 @ [31:16] (td0 hi16=0)
                 (int)(140u << 16),  // tile_dim0 @ [31:16] (td1 hi16=0)
                 32,                 // tile_dim1=32, tile_dim2=0
                 14000,              // tensor_dim0_stride low32 (elements)
                 0, 0 };
  int32x4 gz = {0, 0, 0, 0};
#if defined(__clang_major__) && __clang_major__ >= 23
  int32x8 gz8 = {0, 0, 0, 0, 0, 0, 0, 0};
  __builtin_amdgcn_tensor_load_to_lds(g0, g1, gz, gz, gz8, 0);
#else
  __builtin_amdgcn_tensor_load_to_lds(g0, g1, gz, gz, 0);
#endif
}
#endif

// --------------------------- prologue kernels ------------------------------

__global__ void pack_w(const float* __restrict__ W, u16* __restrict__ dst,
                       int NT, int KT, int srcK) {
  int gid = blockIdx.x * blockDim.x + threadIdx.x;
  int total = NT * KT * 32;
  if (gid >= total) return;
  int lane = gid & 31, tile = gid >> 5;
  int nt = tile / KT, kt = tile - nt * KT;
  int n = lane & 15, khalf = lane >> 4;
  const float* row = W + (size_t)(nt * 16 + n) * srcK;
  u16* d = dst + (size_t)tile * TILE_ELEMS + lane * 16;
#pragma unroll
  for (int j = 0; j < 16; ++j) {
    int k = kt * 32 + khalf * 8 + (j < 8 ? j : j + 8);
    d[j] = f2bf(k < srcK ? row[k] : 0.f);
  }
}

__global__ void make_bias(const float* __restrict__ bi,
                          const float* __restrict__ bh,
                          float* __restrict__ dst, int n) {
  int i = blockIdx.x * blockDim.x + threadIdx.x;
  if (i < n) dst[i] = bi[i] + bh[i];
}

// ------------------------------ main kernel --------------------------------
// LDS (single static block, offsets known for the TDM descriptor):
//   XF : 2 x (32x140 f32)  x_t staging (TDM ping-pong)          @      0
//   HB : 32 x 1032 bf16    [h0 ping|h0 pong|h1 ping|h1 pong]    @  35840
//   XB : 32 x 264 bf16     x_t(bf16, cols<160) / relu-a1(<256)  @ 101888
//   RED: 8 x 32 f32        final-dot partials                   @ 118784

#define SH 1032
#define SX 264
#define XF_BYTES (32 * IN_DIM * 4)              // 17920
#define HB_OFF   (2 * XF_BYTES)                 // 35840
#define XB_OFF   (HB_OFF + 32 * SH * 2)         // 101888
#define RED_OFF  (XB_OFF + 32 * SX * 2)         // 118784
#define SMEM_TOTAL (RED_OFF + 8 * 32 * 4)       // 119808

__global__ __launch_bounds__(256)
void lstm_main(const float* __restrict__ x,
               const u16* __restrict__ W0x, const u16* __restrict__ W0h,
               const u16* __restrict__ W1x, const u16* __restrict__ W1h,
               const u16* __restrict__ Wa1,
               const float* __restrict__ b0v, const float* __restrict__ b1v,
               const float* __restrict__ ba1, const float* __restrict__ wa2,
               const float* __restrict__ ba2, float* __restrict__ out) {
  __shared__ __align__(16) char smem[SMEM_TOTAL];
  float* XF  = (float*)smem;
  u16*   HB  = (u16*)(smem + HB_OFF);
  u16*   XB  = (u16*)(smem + XB_OFF);
  float* RED = (float*)(smem + RED_OFF);

  const int tid  = threadIdx.x;
  const int lane = tid & 31;
  const int wv   = tid >> 5;          // wave id, 0..7
  const int m    = lane & 15;
  const int kh   = lane >> 4;
  const int hi8  = kh * 8;            // C-layout: lanes 16..31 hold M=r+8
  const int n16  = lane & 15;
  const int rowBlk = blockIdx.x * 32;

  float c0s[2][16], c1s[2][16];       // cell state: [chunk][mtile*8+r]
#pragma unroll
  for (int i = 0; i < 2; ++i)
#pragma unroll
    for (int j = 0; j < 16; ++j) { c0s[i][j] = 0.f; c1s[i][j] = 0.f; }

  for (int idx = tid; idx < 32 * 1024; idx += 256) {
    int r = idx >> 10, c = idx & 1023;
    HB[r * SH + c] = 0;
  }

#if USE_TDM
  if (wv == 0)   // request x_0 before the first barrier; TENSORcnt tracks it
    tdm_load_x(x + ((size_t)rowBlk * T_LEN + 0) * IN_DIM, 0u);
#endif
  __syncthreads();

  const v8f vzero = {0.f, 0.f, 0.f, 0.f, 0.f, 0.f, 0.f, 0.f};

  for (int t = 0; t < T_LEN; ++t) {
    const int p = t & 1;

    // ---- stage x_t into XB as bf16 (pad cols 140..159 zeroed) ----
#if USE_TDM
    __builtin_amdgcn_s_wait_tensorcnt(0);   // x_t tile landed in XF[p]
    __syncthreads();                        // LDS visibility across waves
    {
      const float* xf = XF + p * (32 * IN_DIM);
      for (int idx = tid; idx < 32 * IN_PAD; idx += 256) {
        int r = idx / IN_PAD, k = idx - r * IN_PAD;
        float v = (k < IN_DIM) ? xf[r * IN_DIM + k] : 0.f;
        XB[r * SX + k] = f2bf(v);
      }
    }
    if (wv == 0 && t + 1 < T_LEN)           // async prefetch of x_{t+1}
      tdm_load_x(x + ((size_t)rowBlk * T_LEN + (t + 1)) * IN_DIM,
                 (unsigned)((p ^ 1) * XF_BYTES));
#else
    for (int idx = tid; idx < 32 * IN_PAD; idx += 256) {
      int r = idx / IN_PAD, k = idx - r * IN_PAD;
      float v = 0.f;
      if (k < IN_DIM) v = x[((size_t)(rowBlk + r) * T_LEN + t) * IN_DIM + k];
      XB[r * SX + k] = f2bf(v);
    }
#endif
    __syncthreads();

    const int w0 = p * 256,       r0 = 256 - w0;             // h0 slots
    const int w1 = 512 + p * 256, r1 = 512 + 256 - p * 256;  // h1 slots

    // =================== layer 0: g = [x|h0] @ W^T ===================
#pragma unroll
    for (int cc = 0; cc < 2; ++cc) {
      v8f acc[2][4];
#pragma unroll
      for (int mt = 0; mt < 2; ++mt)
#pragma unroll
        for (int g = 0; g < 4; ++g) acc[mt][g] = vzero;

#pragma unroll 1
      for (int kt = 0; kt < IN_PAD / 32; ++kt) {       // x part, K=160
        v16bf a0 = load_a(XB, SX, 0,  kt * 32, m, kh);
        v16bf a1 = load_a(XB, SX, 16, kt * 32, m, kh);
#pragma unroll
        for (int g = 0; g < 4; ++g) {
          v16bf b = load_b(W0x, (g * 16 + wv * 2 + cc) * 5 + kt, lane);
          acc[0][g] = wmma_bf16(a0, b, acc[0][g]);
          acc[1][g] = wmma_bf16(a1, b, acc[1][g]);
        }
      }
#pragma unroll 1
      for (int kt = 0; kt < 8; ++kt) {                 // h0 part, K=256
        v16bf a0 = load_a(HB, SH, 0,  r0 + kt * 32, m, kh);
        v16bf a1 = load_a(HB, SH, 16, r0 + kt * 32, m, kh);
#pragma unroll
        for (int g = 0; g < 4; ++g) {
          v16bf b = load_b(W0h, (g * 16 + wv * 2 + cc) * 8 + kt, lane);
          acc[0][g] = wmma_bf16(a0, b, acc[0][g]);
          acc[1][g] = wmma_bf16(a1, b, acc[1][g]);
        }
      }
      const int colBase = wv * 32 + cc * 16;
      const float bi = b0v[0 * HID + colBase + n16];
      const float bf = b0v[1 * HID + colBase + n16];
      const float bg = b0v[2 * HID + colBase + n16];
      const float bo = b0v[3 * HID + colBase + n16];
#pragma unroll
      for (int mt = 0; mt < 2; ++mt)
#pragma unroll
        for (int r = 0; r < 8; ++r) {
          float iv = sigm(acc[mt][0][r] + bi);
          float fv = sigm(acc[mt][1][r] + bf);
          float gv = tanh_(acc[mt][2][r] + bg);
          float ov = sigm(acc[mt][3][r] + bo);
          float c  = fv * c0s[cc][mt * 8 + r] + iv * gv;
          c0s[cc][mt * 8 + r] = c;
          float h  = ov * tanh_(c);
          int row = mt * 16 + hi8 + r;
          HB[row * SH + w0 + colBase + n16] = f2bf(h);
        }
    }
    __syncthreads();

    // =================== layer 1: g = [h0_new|h1] @ W^T ===================
#pragma unroll
    for (int cc = 0; cc < 2; ++cc) {
      v8f acc[2][4];
#pragma unroll
      for (int mt = 0; mt < 2; ++mt)
#pragma unroll
        for (int g = 0; g < 4; ++g) acc[mt][g] = vzero;

#pragma unroll 1
      for (int kt = 0; kt < 8; ++kt) {                 // h0_new part
        v16bf a0 = load_a(HB, SH, 0,  w0 + kt * 32, m, kh);
        v16bf a1 = load_a(HB, SH, 16, w0 + kt * 32, m, kh);
#pragma unroll
        for (int g = 0; g < 4; ++g) {
          v16bf b = load_b(W1x, (g * 16 + wv * 2 + cc) * 8 + kt, lane);
          acc[0][g] = wmma_bf16(a0, b, acc[0][g]);
          acc[1][g] = wmma_bf16(a1, b, acc[1][g]);
        }
      }
#pragma unroll 1
      for (int kt = 0; kt < 8; ++kt) {                 // h1_old part
        v16bf a0 = load_a(HB, SH, 0,  r1 + kt * 32, m, kh);
        v16bf a1 = load_a(HB, SH, 16, r1 + kt * 32, m, kh);
#pragma unroll
        for (int g = 0; g < 4; ++g) {
          v16bf b = load_b(W1h, (g * 16 + wv * 2 + cc) * 8 + kt, lane);
          acc[0][g] = wmma_bf16(a0, b, acc[0][g]);
          acc[1][g] = wmma_bf16(a1, b, acc[1][g]);
        }
      }
      const int colBase = wv * 32 + cc * 16;
      const float bi = b1v[0 * HID + colBase + n16];
      const float bf = b1v[1 * HID + colBase + n16];
      const float bg = b1v[2 * HID + colBase + n16];
      const float bo = b1v[3 * HID + colBase + n16];
#pragma unroll
      for (int mt = 0; mt < 2; ++mt)
#pragma unroll
        for (int r = 0; r < 8; ++r) {
          float iv = sigm(acc[mt][0][r] + bi);
          float fv = sigm(acc[mt][1][r] + bf);
          float gv = tanh_(acc[mt][2][r] + bg);
          float ov = sigm(acc[mt][3][r] + bo);
          float c  = fv * c1s[cc][mt * 8 + r] + iv * gv;
          c1s[cc][mt * 8 + r] = c;
          float h  = ov * tanh_(c);
          int row = mt * 16 + hi8 + r;
          HB[row * SH + w1 + colBase + n16] = f2bf(h);
        }
    }
    __syncthreads();

    // ============ MLP: relu(h1 @ Wa1^T + ba1) -> XB (a_mid) ============
    {
      v8f acc[2][2];
#pragma unroll
      for (int mt = 0; mt < 2; ++mt)
#pragma unroll
        for (int j = 0; j < 2; ++j) acc[mt][j] = vzero;

#pragma unroll 1
      for (int kt = 0; kt < 8; ++kt) {
        v16bf a0 = load_a(HB, SH, 0,  w1 + kt * 32, m, kh);
        v16bf a1 = load_a(HB, SH, 16, w1 + kt * 32, m, kh);
#pragma unroll
        for (int j = 0; j < 2; ++j) {
          v16bf b = load_b(Wa1, (wv * 2 + j) * 8 + kt, lane);
          acc[0][j] = wmma_bf16(a0, b, acc[0][j]);
          acc[1][j] = wmma_bf16(a1, b, acc[1][j]);
        }
      }
#pragma unroll
      for (int j = 0; j < 2; ++j) {
        int col = (wv * 2 + j) * 16 + n16;
        float bb = ba1[col];
#pragma unroll
        for (int mt = 0; mt < 2; ++mt)
#pragma unroll
          for (int r = 0; r < 8; ++r) {
            float v = acc[mt][j][r] + bb;
            v = v > 0.f ? v : 0.f;
            int row = mt * 16 + hi8 + r;
            XB[row * SX + col] = f2bf(v);
          }
      }
    }
    __syncthreads();

    // =================== final dot with W_a2 + b_a2 ===================
    {
      int r = tid & 31, seg = tid >> 5;       // 8 segments x 32 cols
      float s = 0.f;
#pragma unroll 4
      for (int k = seg * 32; k < seg * 32 + 32; ++k)
        s += bf2f(XB[r * SX + k]) * wa2[k];
      RED[seg * 32 + r] = s;
      __syncthreads();
      if (tid < 32) {
        float o = ba2[0];
#pragma unroll
        for (int sgi = 0; sgi < 8; ++sgi) o += RED[sgi * 32 + tid];
        out[(size_t)t * B_TOT + rowBlk + tid] = o;   // t-major flatten
      }
    }
    __syncthreads();
  }
}

// ------------------------------- launcher ----------------------------------

extern "C" void kernel_launch(void* const* d_in, const int* in_sizes, int n_in,
                              void* d_out, int out_size, void* d_ws, size_t ws_size,
                              hipStream_t stream) {
  const float* x     = (const float*)d_in[0];
  const float* W_ih0 = (const float*)d_in[1];
  const float* W_hh0 = (const float*)d_in[2];
  const float* b_ih0 = (const float*)d_in[3];
  const float* b_hh0 = (const float*)d_in[4];
  const float* W_ih1 = (const float*)d_in[5];
  const float* W_hh1 = (const float*)d_in[6];
  const float* b_ih1 = (const float*)d_in[7];
  const float* b_hh1 = (const float*)d_in[8];
  const float* W_a1  = (const float*)d_in[9];
  const float* b_a1  = (const float*)d_in[10];
  const float* W_a2  = (const float*)d_in[11];
  const float* b_a2  = (const float*)d_in[12];

  u16* ws  = (u16*)d_ws;
  u16* W0x = ws;                                   // 64 x 5 tiles
  u16* W0h = W0x + (size_t)64 * 5 * TILE_ELEMS;    // 64 x 8 tiles
  u16* W1x = W0h + (size_t)64 * 8 * TILE_ELEMS;
  u16* W1h = W1x + (size_t)64 * 8 * TILE_ELEMS;
  u16* Wa1 = W1h + (size_t)64 * 8 * TILE_ELEMS;    // 16 x 8 tiles
  float* b0 = (float*)(Wa1 + (size_t)16 * 8 * TILE_ELEMS);
  float* b1 = b0 + 1024;

  pack_w<<<(64 * 5 * 32 + 255) / 256, 256, 0, stream>>>(W_ih0, W0x, 64, 5, IN_DIM);
  pack_w<<<(64 * 8 * 32 + 255) / 256, 256, 0, stream>>>(W_hh0, W0h, 64, 8, HID);
  pack_w<<<(64 * 8 * 32 + 255) / 256, 256, 0, stream>>>(W_ih1, W1x, 64, 8, HID);
  pack_w<<<(64 * 8 * 32 + 255) / 256, 256, 0, stream>>>(W_hh1, W1h, 64, 8, HID);
  pack_w<<<(16 * 8 * 32 + 255) / 256, 256, 0, stream>>>(W_a1,  Wa1, 16, 8, HID);
  make_bias<<<4, 256, 0, stream>>>(b_ih0, b_hh0, b0, 1024);
  make_bias<<<4, 256, 0, stream>>>(b_ih1, b_hh1, b1, 1024);

  lstm_main<<<B_TOT / 32, 256, 0, stream>>>(
      x, W0x, W0h, W1x, W1h, Wa1, b0, b1, b_a1, W_a2, b_a2, (float*)d_out);
}
